// RepetitionAttention_80152679678781
// MI455X (gfx1250) — matrix operand order repaired and checked by reading, not compiled, not run on hardware
//
#include <hip/hip_runtime.h>
#include <math.h>

// Problem constants (from the reference)
#define B_  2
#define S_  2048
#define D_  1024
#define H_  16
#define HD_ 64

typedef __bf16 bf16_t;
typedef __attribute__((ext_vector_type(16))) __bf16 v16bf;
typedef __attribute__((ext_vector_type(8)))  __bf16 v8bf;
typedef __attribute__((ext_vector_type(8)))  float  v8f;

// ---------------------------------------------------------------------------
// helpers
// ---------------------------------------------------------------------------
__device__ inline v8f vzero8() {
  v8f z;
#pragma unroll
  for (int i = 0; i < 8; ++i) z[i] = 0.0f;
  return z;
}

__device__ inline v8f wmma_bf16(v16bf a, v16bf b, v8f c) {
  // D = A(16x32 bf16) x B(32x16 bf16) + C(16x16 f32)
  return __builtin_amdgcn_wmma_f32_16x16x32_bf16(
      /*neg_a=*/false, a, /*neg_b=*/false, b,
      /*c_mod=*/(short)0, c, /*reuse_a=*/false, /*reuse_b=*/false);
}

// A-fragment: 16x32 bf16 tile from row-major [M][K] source.
// Documented layout: lane&15 = M row; half-wave adds K+8; vgpr4..7 hold K+16..23.
__device__ inline v16bf ld_afrag(const bf16_t* __restrict__ base, int lda,
                                 int m0, int k0, int lane) {
  const int m  = m0 + (lane & 15);
  const int kk = k0 + ((lane >> 4) << 3);
  const bf16_t* p = base + (size_t)m * lda + kk;
  v8bf lo = *(const v8bf*)(p);       // K = kk .. kk+7
  v8bf hi = *(const v8bf*)(p + 16);  // K = kk+16 .. kk+23
  v16bf a;
#pragma unroll
  for (int i = 0; i < 8; ++i) { a[i] = lo[i]; a[i + 8] = hi[i]; }
  return a;
}

// B-fragment: 32x16 bf16 tile; source is "transposed" [N][K] row-major so each
// lane (column N = n0 + lane&15) reads K = kk..kk+7 and kk+16..kk+23; half-wave = K+8.
__device__ inline v16bf ld_bfrag(const bf16_t* __restrict__ base, int ldb,
                                 int n0, int k0, int lane) {
  const int n  = n0 + (lane & 15);
  const int kk = k0 + ((lane >> 4) << 3);
  const bf16_t* p = base + (size_t)n * ldb + kk;
  v8bf lo = *(const v8bf*)(p);
  v8bf hi = *(const v8bf*)(p + 16);
  v16bf b;
#pragma unroll
  for (int i = 0; i < 8; ++i) { b[i] = lo[i]; b[i + 8] = hi[i]; }
  return b;
}

__device__ inline float rowmax16(float v) {
#pragma unroll
  for (int off = 8; off; off >>= 1) v = fmaxf(v, __shfl_xor(v, off, 16));
  return v;
}
__device__ inline float rowsum16(float v) {
#pragma unroll
  for (int off = 8; off; off >>= 1) v += __shfl_xor(v, off, 16);
  return v;
}

// ---------------------------------------------------------------------------
// prep kernels: fp32 -> bf16, row norms, weight transpose
// ---------------------------------------------------------------------------
__global__ __launch_bounds__(256) void prep_x_kernel(
    const float* __restrict__ x, bf16_t* __restrict__ xb, bf16_t* __restrict__ xnb) {
  __shared__ float red[256];
  const int row = blockIdx.x;
  const float* xr = x + (size_t)row * D_;
  float ss = 0.0f;
  for (int t = threadIdx.x; t < D_; t += 256) { float v = xr[t]; ss += v * v; }
  red[threadIdx.x] = ss;
  __syncthreads();
  for (int s = 128; s > 0; s >>= 1) {
    if (threadIdx.x < s) red[threadIdx.x] += red[threadIdx.x + s];
    __syncthreads();
  }
  const float inv = 1.0f / fmaxf(sqrtf(red[0]), 1e-12f);  // torch F.normalize eps
  for (int t = threadIdx.x; t < D_; t += 256) {
    float v = xr[t];
    xb [(size_t)row * D_ + t] = (bf16_t)v;
    xnb[(size_t)row * D_ + t] = (bf16_t)(v * inv);
  }
}

__global__ __launch_bounds__(256) void prep_w_kernel(
    const float* __restrict__ W, bf16_t* __restrict__ WT) {
  const int idx = blockIdx.x * 256 + threadIdx.x;   // grid covers D*D exactly
  const int n = idx >> 10;          // idx / D_
  const int k = idx & (D_ - 1);     // idx % D_
  WT[(size_t)n * D_ + k] = (bf16_t)W[(size_t)k * D_ + n];
}

// ---------------------------------------------------------------------------
// WMMA GEMM:  C[M,N] = A[M,K] * BT[N,K]^T + bias
// One wave computes a 16x64 macro-tile (4 accumulators): per k-step one
// A-fragment is reused by 4 WMMAs (A traffic /4, 1.25 loads per wmma).
// mode 0: bf16 out as Q/K layout  [B,H,S,HD]
// mode 2: bf16 out as V^T layout  [B,H,HD,S]
// mode 3: f32  out row-major      [M,N]
// Grids divide exactly (no partial waves: WMMA requires EXEC == all ones).
// ---------------------------------------------------------------------------
__global__ __launch_bounds__(256) void gemm_bf16_kernel(
    const bf16_t* __restrict__ A, const bf16_t* __restrict__ BT,
    const float* __restrict__ bias,
    bf16_t* __restrict__ out_bf, float* __restrict__ out_f,
    int M, int N, int K, int mode) {
  const int lane = threadIdx.x & 31;
  const int wave = threadIdx.x >> 5;
  const int ntn  = N >> 6;                       // 64-wide macro tiles
  const int tile = blockIdx.x * 8 + wave;
  if (tile >= (M >> 4) * ntn) return;            // wave-uniform
  const int tm = (tile / ntn) << 4;
  const int tn = (tile % ntn) << 6;

  v8f acc0 = vzero8(), acc1 = vzero8(), acc2 = vzero8(), acc3 = vzero8();
  for (int k0 = 0; k0 < K; k0 += 32) {
    __builtin_prefetch(A  + (size_t)tm * K + k0 + 64, 0, 1);
    __builtin_prefetch(BT + (size_t)tn * K + k0 + 64, 0, 1);
    v16bf a  = ld_afrag(A, K, tm, k0, lane);
    v16bf b0 = ld_bfrag(BT, K, tn,      k0, lane);
    v16bf b1 = ld_bfrag(BT, K, tn + 16, k0, lane);
    v16bf b2 = ld_bfrag(BT, K, tn + 32, k0, lane);
    v16bf b3 = ld_bfrag(BT, K, tn + 48, k0, lane);
    acc0 = wmma_bf16(a, b0, acc0);
    acc1 = wmma_bf16(a, b1, acc1);
    acc2 = wmma_bf16(a, b2, acc2);
    acc3 = wmma_bf16(a, b3, acc3);
  }

  v8f accs[4] = {acc0, acc1, acc2, acc3};
#pragma unroll
  for (int j = 0; j < 4; ++j) {
    const int   n  = tn + (j << 4) + (lane & 15);
    const float bb = bias ? bias[n] : 0.0f;
#pragma unroll
    for (int i = 0; i < 8; ++i) {
      const int   m = tm + i + ((lane >> 4) << 3);  // C layout: vgpr i, half-wave -> M+8
      const float v = accs[j][i] + bb;
      if (mode == 3) {
        out_f[(size_t)m * N + n] = v;
      } else {
        const int b_ = m / S_, s = m % S_;
        const int h  = n / HD_, hd = n % HD_;
        size_t idx;
        if (mode == 2) idx = (((size_t)b_ * H_ + h) * HD_ + hd) * S_ + s;   // V^T
        else           idx = (((size_t)b_ * H_ + h) * S_  + s ) * HD_ + hd; // Q / K
        out_bf[idx] = (bf16_t)v;
      }
    }
  }
}

// ---------------------------------------------------------------------------
// Repetition scores: rep[b,s] = max_{k != s} cos(x_s, x_k)
// Reference zeroes the diagonal (sim * (1-eye)), so diag contributes 0.
// One wave owns a 16-row strip; key tiles processed 4 at a time so the
// A-fragment of the strip is reused by 4 WMMAs per k-step.
// ---------------------------------------------------------------------------
__global__ __launch_bounds__(256) void rep_kernel(
    const bf16_t* __restrict__ xn, float* __restrict__ rep_out) {
  const int lane  = threadIdx.x & 31;
  const int wave  = threadIdx.x >> 5;
  const int strip = blockIdx.x * 8 + wave;       // B*(S/16) strips, exact
  const int b  = strip / (S_ >> 4);
  const int tm = (strip % (S_ >> 4)) << 4;
  const bf16_t* X = xn + (size_t)b * S_ * D_;

  float rowmax[8];
#pragma unroll
  for (int i = 0; i < 8; ++i) rowmax[i] = -1e30f;

  for (int tn = 0; tn < S_; tn += 64) {
    v8f acc0 = vzero8(), acc1 = vzero8(), acc2 = vzero8(), acc3 = vzero8();
    for (int k0 = 0; k0 < D_; k0 += 32) {
      v16bf a  = ld_afrag(X, D_, tm, k0, lane);
      v16bf b0 = ld_bfrag(X, D_, tn,      k0, lane);  // xn rows are [N][K] form
      v16bf b1 = ld_bfrag(X, D_, tn + 16, k0, lane);
      v16bf b2 = ld_bfrag(X, D_, tn + 32, k0, lane);
      v16bf b3 = ld_bfrag(X, D_, tn + 48, k0, lane);
      acc0 = wmma_bf16(a, b0, acc0);
      acc1 = wmma_bf16(a, b1, acc1);
      acc2 = wmma_bf16(a, b2, acc2);
      acc3 = wmma_bf16(a, b3, acc3);
    }
    v8f accs[4] = {acc0, acc1, acc2, acc3};
#pragma unroll
    for (int j = 0; j < 4; ++j) {
      const int n = tn + (j << 4) + (lane & 15);
#pragma unroll
      for (int i = 0; i < 8; ++i) {
        const int m = tm + i + ((lane >> 4) << 3);
        const float v = (m == n) ? 0.0f : accs[j][i];  // diagonal -> 0 (matches ref)
        rowmax[i] = fmaxf(rowmax[i], v);
      }
    }
  }
#pragma unroll
  for (int i = 0; i < 8; ++i) rowmax[i] = rowmax16(rowmax[i]);
  if ((lane & 15) == 0) {
#pragma unroll
    for (int i = 0; i < 8; ++i) {
      const int m = tm + i + ((lane >> 4) << 3);
      rep_out[(size_t)b * S_ + m] = rowmax[i];
    }
  }
}

// ---------------------------------------------------------------------------
// Flash attention: one wave per (b, h, 16-query tile), 32 keys per step.
// The per-query rep bonus is constant along keys -> softmax-invariant -> omitted.
// P (C layout) is bounced through LDS to re-swizzle into an A fragment.
// ---------------------------------------------------------------------------
__global__ __launch_bounds__(256) void attn_kernel(
    const bf16_t* __restrict__ Q, const bf16_t* __restrict__ Km,
    const bf16_t* __restrict__ Vt, const unsigned char* __restrict__ mask,
    bf16_t* __restrict__ attn) {
  __shared__ bf16_t pbuf[8][16 * 32];
  const int lane   = threadIdx.x & 31;
  const int wave   = threadIdx.x >> 5;
  const int qtiles = S_ >> 4;
  const int gw = blockIdx.x * 8 + wave;          // B*H*qtiles waves, exact
  const int tq = (gw % qtiles) << 4;
  const int h  = (gw / qtiles) % H_;
  const int b  = gw / (qtiles * H_);

  const bf16_t* Qh = Q  + (((size_t)b * H_ + h) * S_)  * HD_;
  const bf16_t* Kh = Km + (((size_t)b * H_ + h) * S_)  * HD_;
  const bf16_t* Vh = Vt + (((size_t)b * H_ + h) * HD_) * S_;

  const v16bf aq0 = ld_afrag(Qh, HD_, tq, 0,  lane);
  const v16bf aq1 = ld_afrag(Qh, HD_, tq, 32, lane);

  float mrow[8], lrow[8];
#pragma unroll
  for (int i = 0; i < 8; ++i) { mrow[i] = -1e30f; lrow[i] = 0.0f; }
  v8f o0 = vzero8(), o1 = vzero8(), o2 = vzero8(), o3 = vzero8();

  bf16_t* myp = &pbuf[wave][0];
  const int c0 = lane & 15;
  const float scl = 0.125f;                      // 1/sqrt(HD)

  for (int kb = 0; kb < S_; kb += 32) {
    // scores: two 16x16 tiles (keys kb..kb+15 and kb+16..kb+31)
    v8f s0 = vzero8(), s1 = vzero8();
    {
      v16bf bk;
      bk = ld_bfrag(Kh, HD_, kb,      0,  lane); s0 = wmma_bf16(aq0, bk, s0);
      bk = ld_bfrag(Kh, HD_, kb,      32, lane); s0 = wmma_bf16(aq1, bk, s0);
      bk = ld_bfrag(Kh, HD_, kb + 16, 0,  lane); s1 = wmma_bf16(aq0, bk, s1);
      bk = ld_bfrag(Kh, HD_, kb + 16, 32, lane); s1 = wmma_bf16(aq1, bk, s1);
    }
    const bool ok0 = mask[(size_t)b * S_ + kb +      c0] != 0;
    const bool ok1 = mask[(size_t)b * S_ + kb + 16 + c0] != 0;
#pragma unroll
    for (int i = 0; i < 8; ++i) {
      s0[i] = ok0 ? s0[i] * scl : -1e30f;        // per-lane select, no divergence
      s1[i] = ok1 ? s1[i] * scl : -1e30f;
    }
    // online softmax update + stage P into LDS (bf16)
#pragma unroll
    for (int i = 0; i < 8; ++i) {
      float t  = rowmax16(fmaxf(s0[i], s1[i]));
      float nm = fmaxf(mrow[i], t);
      float corr = __expf(mrow[i] - nm);
      float p0 = __expf(s0[i] - nm);
      float p1 = __expf(s1[i] - nm);
      lrow[i] = lrow[i] * corr + rowsum16(p0 + p1);
      mrow[i] = nm;
      o0[i] *= corr; o1[i] *= corr; o2[i] *= corr; o3[i] *= corr;
      const int r = i + ((lane >> 4) << 3);      // C-layout row
      myp[r * 32 +      c0] = (bf16_t)p0;
      myp[r * 32 + 16 + c0] = (bf16_t)p1;
    }
    asm volatile("s_wait_dscnt 0" ::: "memory"); // intra-wave LDS RAW fence
    // re-swizzle P (16x32) into an A fragment
    v16bf ap;
    {
      const int m  = lane & 15;
      const int ko = (lane >> 4) << 3;
      v8bf lo = *(const v8bf*)&myp[m * 32 + ko];
      v8bf hi = *(const v8bf*)&myp[m * 32 + ko + 16];
#pragma unroll
      for (int i = 0; i < 8; ++i) { ap[i] = lo[i]; ap[i + 8] = hi[i]; }
    }
    // O += P * V  (V^T[hd][s] is directly a B operand)
    o0 = wmma_bf16(ap, ld_bfrag(Vh, S_, 0,  kb, lane), o0);
    o1 = wmma_bf16(ap, ld_bfrag(Vh, S_, 16, kb, lane), o1);
    o2 = wmma_bf16(ap, ld_bfrag(Vh, S_, 32, kb, lane), o2);
    o3 = wmma_bf16(ap, ld_bfrag(Vh, S_, 48, kb, lane), o3);
  }

#pragma unroll
  for (int i = 0; i < 8; ++i) {
    const int   s   = tq + i + ((lane >> 4) << 3);
    const float inv = 1.0f / lrow[i];
    bf16_t* dst = attn + ((size_t)b * S_ + s) * D_ + h * HD_;
    dst[c0]      = (bf16_t)(o0[i] * inv);
    dst[16 + c0] = (bf16_t)(o1[i] * inv);
    dst[32 + c0] = (bf16_t)(o2[i] * inv);
    dst[48 + c0] = (bf16_t)(o3[i] * inv);
  }
}

// ---------------------------------------------------------------------------
// launch
// ---------------------------------------------------------------------------
extern "C" void kernel_launch(void* const* d_in, const int* in_sizes, int n_in,
                              void* d_out, int out_size, void* d_ws, size_t ws_size,
                              hipStream_t stream) {
  const float* x  = (const float*)d_in[0];
  const unsigned char* mask = (const unsigned char*)d_in[1];
  const float* Wq = (const float*)d_in[2]; const float* bq = (const float*)d_in[3];
  const float* Wk = (const float*)d_in[4]; const float* bk = (const float*)d_in[5];
  const float* Wv = (const float*)d_in[6]; const float* bv = (const float*)d_in[7];
  const float* Wo = (const float*)d_in[8]; const float* bo = (const float*)d_in[9];

  float* out = (float*)d_out;                       // [B,S,D] f32
  float* rep = out + (size_t)B_ * S_ * D_;          // [B,S]   f32

  char* p = (char*)d_ws;
  const size_t szX = (size_t)B_ * S_ * D_ * sizeof(bf16_t);   // 8 MB each
  const size_t szW = (size_t)D_ * D_ * sizeof(bf16_t);        // 2 MB each
  bf16_t* xb   = (bf16_t*)p; p += szX;
  bf16_t* xnb  = (bf16_t*)p; p += szX;
  bf16_t* WqT  = (bf16_t*)p; p += szW;
  bf16_t* WkT  = (bf16_t*)p; p += szW;
  bf16_t* WvT  = (bf16_t*)p; p += szW;
  bf16_t* WoT  = (bf16_t*)p; p += szW;
  bf16_t* Qb   = (bf16_t*)p; p += szX;
  bf16_t* Kb   = (bf16_t*)p; p += szX;
  bf16_t* Vt   = (bf16_t*)p; p += szX;
  bf16_t* attn = (bf16_t*)p; p += szX;              // total ~56 MB

  prep_x_kernel<<<B_ * S_, 256, 0, stream>>>(x, xb, xnb);
  prep_w_kernel<<<(D_ * D_) / 256, 256, 0, stream>>>(Wq, WqT);
  prep_w_kernel<<<(D_ * D_) / 256, 256, 0, stream>>>(Wk, WkT);
  prep_w_kernel<<<(D_ * D_) / 256, 256, 0, stream>>>(Wv, WvT);
  prep_w_kernel<<<(D_ * D_) / 256, 256, 0, stream>>>(Wo, WoT);

  const int M = B_ * S_;
  const int gemm_blocks = ((M >> 4) * (D_ >> 6)) / 8;   // 512 exact (16x64 tiles)
  gemm_bf16_kernel<<<gemm_blocks, 256, 0, stream>>>(xb, WqT, bq, Qb, nullptr, M, D_, D_, 0);
  gemm_bf16_kernel<<<gemm_blocks, 256, 0, stream>>>(xb, WkT, bk, Kb, nullptr, M, D_, D_, 0);
  gemm_bf16_kernel<<<gemm_blocks, 256, 0, stream>>>(xb, WvT, bv, Vt, nullptr, M, D_, D_, 2);

  rep_kernel<<<(B_ * (S_ >> 4)) / 8, 256, 0, stream>>>(xnb, rep);

  attn_kernel<<<(B_ * H_ * (S_ >> 4)) / 8, 256, 0, stream>>>(Qb, Kb, Vt, mask, attn);

  gemm_bf16_kernel<<<gemm_blocks, 256, 0, stream>>>(attn, WoT, bo, nullptr, out, M, D_, D_, 3);
}
